// FullyConnectedTensorProduct_5016521801922
// MI455X (gfx1250) — matrix operand order repaired and checked by reading, not compiled, not run on hardware
//
#include <hip/hip_runtime.h>

// ---------------- problem constants ----------------
#define BATCH_N        (1u << 20)
#define ITEMS_PER_WAVE 32
#define WAVES_PER_BLK  2
#define ITEMS_PER_BLK  (ITEMS_PER_WAVE * WAVES_PER_BLK)   // 64
#define S_ITEM         1040                                // bytes per item region (4*256 + 16 pad)
#define WAVE_LDS       (ITEMS_PER_WAVE * S_ITEM)           // 33280
#define GRID_BLOCKS    2048

typedef __attribute__((ext_vector_type(16))) _Float16 v16h;
typedef __attribute__((ext_vector_type(8)))  _Float16 v8h;
typedef __attribute__((ext_vector_type(2)))  __fp16   pkh2;   // cvt_pkrtz return type
typedef __attribute__((ext_vector_type(8)))  float    v8f;

__device__ __forceinline__ unsigned pk2(float a, float b) {
  union { pkh2 h; unsigned u; } z;
  z.h = __builtin_amdgcn_cvt_pkrtz(a, b);     // v_cvt_pk_rtz_f16_f32
  return z.u;
}

__device__ __forceinline__ void store8h(void* dst, const float* f) {
  uint4 q;
  q.x = pk2(f[0], f[1]);
  q.y = pk2(f[2], f[3]);
  q.z = pk2(f[4], f[5]);
  q.w = pk2(f[6], f[7]);
  *(uint4*)dst = q;
}

__global__ __launch_bounds__(64) void tp_wmma_kernel(const float* __restrict__ x1,
                                                     const float* __restrict__ x2,
                                                     const float* __restrict__ weight,
                                                     float* __restrict__ out)
{
  __shared__ __align__(16) unsigned char lds[WAVES_PER_BLK * WAVE_LDS];

  const int tid  = threadIdx.x;
  const int lane = tid & 31;
  const int wave = tid >> 5;

  // -------- Stage 1: build Wcat[128][16] (f32, scales folded) in LDS --------
  {
    float* wbuf = (float*)lds;
    const float AL = 0.08838834764831845f;            // 1/sqrt(2*8*8)
    const float AI = AL * 0.5773502691896258f;        // AL/sqrt(3)
    for (int idx = tid; idx < 2048; idx += 64) {
      float w = weight[idx];
      int p  = idx >> 9;       // path 0..3 (W0..W3)
      int r  = idx & 511;      // u*64 + v*8 + wcol
      int uv = r >> 3;         // u*8+v : K row within half
      int wc = r & 7;
      int row = uv + ((p & 1) ? 64 : 0);       // W1,W3 in K=64..127
      int col = wc + ((p >= 2) ? 8 : 0);       // W2,W3 in cols 8..15
      wbuf[row * 16 + col] = w * ((p == 1) ? AI : AL);
    }
  }
  __syncthreads();

  // -------- Stage 2: per-lane register-resident B fragments (f16) --------
  // B (32x16): lanes 0-15 hold K=0..7,16..23, lanes 16-31 hold K=8..15,24..31;
  // VGPR r packs consecutive K pairs in that order for column N = lane%16.
  const int ncol = lane & 15;
  const int half = lane >> 4;
  v16h bfrag[4];
  {
    const float* wbuf = (const float*)lds;
#pragma unroll
    for (int j = 0; j < 4; ++j) {
#pragma unroll
      for (int r = 0; r < 8; ++r) {
        int k = 32 * j + 8 * half + 2 * r + (r >= 4 ? 8 : 0);
        bfrag[j][2*r    ] = (_Float16)wbuf[(k    ) * 16 + ncol];
        bfrag[j][2*r + 1] = (_Float16)wbuf[(k + 1) * 16 + ncol];
      }
    }
  }
  __syncthreads();   // wbuf region is reused as wave-0 feature space below

  unsigned char* fw = lds + wave * WAVE_LDS;   // this wave's private LDS region
  const unsigned stride_items = GRID_BLOCKS * ITEMS_PER_BLK;

  // ---- loop-invariant per-lane addresses ----
  // A-fragment base: row M = lane%16 -> item (M>>2), type (M&3); lane-half K split.
  const int i4  = (lane & 15) >> 2;
  const int typ = lane & 3;
  const int klo = half ? 16 : 0;                       // byte offset of K start
  const unsigned char* apbase = fw + i4 * S_ITEM + typ * 256 + klo;

  // D scatter: unconditional stores; invalid lanes hit per-lane trash slot
  // (floats 48..55 of the item region = consumed type-0 feature bytes).
  float* outp[2][4];
  {
    int offs[4];
    offs[0] = (ncol < 8) ? ncol : (48 + (ncol & 7));
#pragma unroll
    for (int kk = 1; kk <= 3; ++kk)
      offs[kk] = (ncol >= 8) ? (8 + (ncol - 8) * 3 + (kk - 1)) : (48 + ncol);
#pragma unroll
    for (int gg = 0; gg < 2; ++gg)
#pragma unroll
      for (int tg = 0; tg < 4; ++tg)
        outp[gg][tg] = (float*)(fw + (gg + 2 * half) * S_ITEM) + offs[tg];
  }

  for (unsigned base = blockIdx.x * ITEMS_PER_BLK; base < BATCH_N; base += stride_items) {
    const unsigned gi = base + wave * ITEMS_PER_WAVE + lane;   // lane == item

    if (base + stride_items < BATCH_N) {                       // global_prefetch
      __builtin_prefetch(x1 + (size_t)(gi + stride_items) * 32, 0, 1);
      __builtin_prefetch(x2 + (size_t)(gi + stride_items) * 32, 0, 1);
    }

    // ---- coalesced 128B/lane input loads ----
    float a1[32], a2[32];
    {
      const float4* p1 = (const float4*)(x1 + (size_t)gi * 32);
      const float4* p2 = (const float4*)(x2 + (size_t)gi * 32);
#pragma unroll
      for (int c = 0; c < 8; ++c) {
        ((float4*)a1)[c] = p1[c];
        ((float4*)a2)[c] = p2[c];
      }
    }
    const float* s1 = a1;
    const float* s2 = a2;
    const float* v1 = a1 + 8;   // v1[u*3+i]
    const float* v2 = a2 + 8;

    // ---- features (f32 math, packed f16) -> LDS in A-matrix K-order ----
    unsigned char* my = fw + lane * S_ITEM;
#pragma unroll
    for (int u = 0; u < 8; ++u) {
      const float su  = s1[u];
      const float vu0 = v1[u*3+0], vu1 = v1[u*3+1], vu2 = v1[u*3+2];
      float t0[8], t1[8], t2a[8], t2b[8], t2c[8], t3a[8], t3b[8], t3c[8];
#pragma unroll
      for (int v = 0; v < 8; ++v) {
        const float w0 = v2[v*3+0], w1 = v2[v*3+1], w2 = v2[v*3+2];
        const float sv = s2[v];
        t0[v]  = su * sv;                              // type0 K<64
        t1[v]  = vu0*w0 + vu1*w1 + vu2*w2;             // type0 K>=64 (1/sqrt3 in W)
        t2a[v] = su * w0;  t2b[v] = su * w1;  t2c[v] = su * w2;   // g_k K<64
        t3a[v] = vu0 * sv; t3b[v] = vu1 * sv; t3c[v] = vu2 * sv;  // g_k K>=64
      }
      store8h(my + 0*256 +       u*16, t0);
      store8h(my + 0*256 + 128 + u*16, t1);
      store8h(my + 1*256 +       u*16, t2a);
      store8h(my + 1*256 + 128 + u*16, t3a);
      store8h(my + 2*256 +       u*16, t2b);
      store8h(my + 2*256 + 128 + u*16, t3b);
      store8h(my + 3*256 +       u*16, t2c);
      store8h(my + 3*256 + 128 + u*16, t3c);
      __builtin_amdgcn_sched_barrier(0);   // cap live ranges / VGPR pressure
    }
    asm volatile("" ::: "memory");
    __builtin_amdgcn_wave_barrier();   // wave-internal; LDS is in-order per wave

    // ---- 8 WMMA tiles, software-pipelined A-fragment loads ----
    v8h fA[8];
#pragma unroll
    for (int j = 0; j < 4; ++j) {
      fA[2*j    ] = *(const v8h*)(apbase + 64*j);
      fA[2*j + 1] = *(const v8h*)(apbase + 64*j + 32);
    }
#pragma unroll
    for (int t = 0; t < 8; ++t) {
      v8h nA[8];
      if (t < 7) {                                   // issue next tile's loads early
        const unsigned char* ap = apbase + (t + 1) * 4 * S_ITEM;
#pragma unroll
        for (int j = 0; j < 4; ++j) {
          nA[2*j    ] = *(const v8h*)(ap + 64*j);
          nA[2*j + 1] = *(const v8h*)(ap + 64*j + 32);
        }
      }
      v8f acc = {0.f, 0.f, 0.f, 0.f, 0.f, 0.f, 0.f, 0.f};
#pragma unroll
      for (int j = 0; j < 4; ++j) {
        v16h am;
#pragma unroll
        for (int e = 0; e < 8; ++e) { am[e] = fA[2*j][e]; am[8+e] = fA[2*j+1][e]; }
        acc = __builtin_amdgcn_wmma_f32_16x16x32_f16(
                  false, am, false, bfrag[j], (short)0, acc, false, false);
      }
      // branch-free D scatter: ds_store_b32 with immediate tile offset only
#pragma unroll
      for (int g = 0; g < 8; ++g)
        outp[g >> 2][g & 3][t * 1040] = acc[g];      // 1040 floats = 4*S_ITEM bytes
      if (t < 7) {
#pragma unroll
        for (int i = 0; i < 8; ++i) fA[i] = nA[i];
      }
    }
    asm volatile("" ::: "memory");
    __builtin_amdgcn_wave_barrier();

    // ---- coalesced 128B/lane output store ----
    {
      float4* po = (float4*)(out + (size_t)gi * 32);
#pragma unroll
      for (int c = 0; c < 8; ++c)
        po[c] = *(const float4*)(fw + lane * S_ITEM + c * 16);
    }
    asm volatile("" ::: "memory");
    __builtin_amdgcn_wave_barrier();   // before next iter overwrites features
  }
}

extern "C" void kernel_launch(void* const* d_in, const int* in_sizes, int n_in,
                              void* d_out, int out_size, void* d_ws, size_t ws_size,
                              hipStream_t stream) {
  const float* x1 = (const float*)d_in[0];
  const float* x2 = (const float*)d_in[1];
  const float* w  = (const float*)d_in[2];
  float* out = (float*)d_out;
  hipLaunchKernelGGL(tp_wmma_kernel, dim3(GRID_BLOCKS), dim3(64), 0, stream,
                     x1, x2, w, out);
}
